// MHASelfAttention_10737418240656
// MI455X (gfx1250) — matrix operand order repaired and checked by reading, not compiled, not run on hardware
//
#include <hip/hip_runtime.h>
#include <hip/hip_bf16.h>
#include <math.h>

#define BB 4
#define SS 2048
#define DD 1024
#define HH 16
#define DK 64

typedef __attribute__((ext_vector_type(16))) __bf16 v16bf;
typedef __attribute__((ext_vector_type(8)))  __bf16 v8bf;
typedef __attribute__((ext_vector_type(8)))  float  v8f;

// A-fragment loader: lane = row, K chunk [16*kg,+16)
static __device__ inline v16bf load_afrag(const __bf16* rowp, int kg) {
  v8bf lo = *(const v8bf*)(rowp + 16 * kg);
  v8bf hi = *(const v8bf*)(rowp + 16 * kg + 8);
  v16bf a;
#pragma unroll
  for (int j = 0; j < 8; ++j) { a[j] = lo[j]; a[8 + j] = hi[j]; }
  return a;
}

// ---------------------------------------------------------------------------
// Kernel 0: fp32 -> bf16 elementwise conversion (bandwidth-bound, one pass)
// ---------------------------------------------------------------------------
__global__ __launch_bounds__(256)
void convert_kernel(const float* __restrict__ in, __bf16* __restrict__ out, int n) {
  const int i = (blockIdx.x * 256 + threadIdx.x) * 4;
  if (i < n) {
    float4 f = *(const float4*)(in + i);
    out[i + 0] = (__bf16)f.x;
    out[i + 1] = (__bf16)f.y;
    out[i + 2] = (__bf16)f.z;
    out[i + 3] = (__bf16)f.w;
  }
}

// ---------------------------------------------------------------------------
// Kernel 1: QKV projection.  out[q,b,h,s,d] = sum_m x[b,s,m] * Wqkv[q,h,d,m]
// One wave computes a 64 (s) x 64 (d) tile: 4 A-frags x 4 B-frags = 16 WMMAs
// per 32-wide k-step. Q,K stored [bh,S,dk]; V stored transposed [bh,dk,S].
// ---------------------------------------------------------------------------
__global__ __launch_bounds__(256)
void qkv_proj_kernel(const __bf16* __restrict__ xb, const __bf16* __restrict__ Wb,
                     __bf16* __restrict__ Qb, __bf16* __restrict__ Kb,
                     __bf16* __restrict__ Vt) {
  const int lane = threadIdx.x & 31;
  const int wave = threadIdx.x >> 5;
  const int gid  = blockIdx.x * 8 + wave;     // 0..6143 = 3*B*H*(S/64)
  const int st   = gid & 31;                  // s tile (64 rows)
  const int bh3  = gid >> 5;                  // 0..191
  const int h    = bh3 & 15;
  const int b    = (bh3 >> 4) & 3;
  const int q    = bh3 >> 6;                  // 0..2
  const int kg   = lane >> 4;
  const int ln   = lane & 15;
  const int s0   = st * 64;

  v8f acc[4][4] = {};                         // [mi][t] 16x16 tiles
  const __bf16* xbase = xb + ((size_t)b * SS + s0 + ln) * DD;
  const __bf16* wbase = Wb + (size_t)(q * HH + h) * DK * DD;

  for (int k0 = 0; k0 < DD; k0 += 32) {
    v16bf a[4];
#pragma unroll
    for (int mi = 0; mi < 4; ++mi)
      a[mi] = load_afrag(xbase + (size_t)mi * 16 * DD + k0, kg);
#pragma unroll
    for (int t = 0; t < 4; ++t) {
      // B fragment (32x16): lane = column d, 16 consecutive K from 16*kg
      v16bf bf = *(const v16bf*)(wbase + (size_t)(t * 16 + ln) * DD + k0 + 16 * kg);
#pragma unroll
      for (int mi = 0; mi < 4; ++mi)
        acc[mi][t] = __builtin_amdgcn_wmma_f32_16x16x32_bf16(
            false, a[mi], false, bf, (short)0, acc[mi][t], false, false);
    }
  }

  // Wave-uniform epilogue select (no per-store divergence / exec games)
  const int bh = b * HH + h;
  if (q < 2) {
    __bf16* dst = (q == 0 ? Qb : Kb) + ((size_t)bh * SS + s0 + 8 * kg) * DK + ln;
#pragma unroll
    for (int mi = 0; mi < 4; ++mi)
#pragma unroll
      for (int t = 0; t < 4; ++t)
#pragma unroll
        for (int r = 0; r < 8; ++r)
          dst[(size_t)(mi * 16 + r) * DK + t * 16] = (__bf16)acc[mi][t][r];
  } else {
    // Vt[(bh*DK + col)*S + row]: r-consecutive rows -> one 16B packed store
#pragma unroll
    for (int mi = 0; mi < 4; ++mi)
#pragma unroll
      for (int t = 0; t < 4; ++t) {
        v8bf pk;
#pragma unroll
        for (int r = 0; r < 8; ++r) pk[r] = (__bf16)acc[mi][t][r];
        *(v8bf*)(Vt + ((size_t)bh * DK + t * 16 + ln) * SS
                    + s0 + mi * 16 + 8 * kg) = pk;
      }
  }
}

// ---------------------------------------------------------------------------
// Kernel 2: causal flash attention per (b,h). One wave owns a 32-row query
// tile (each K/V B-fragment feeds 2 WMMAs); loops over 32-key tiles with
// online softmax; PV accumulated in f32. P goes C-layout -> LDS -> A-layout
// (wave-private staging, s_wait_dscnt only -- wave trip counts diverge).
// ---------------------------------------------------------------------------
__global__ __launch_bounds__(256)
void attn_kernel(const __bf16* __restrict__ Qb, const __bf16* __restrict__ Kb,
                 const __bf16* __restrict__ Vt, __bf16* __restrict__ Cc) {
  __shared__ __bf16 plds[8][32][32];          // 2 KB per wave
  const int lane = threadIdx.x & 31;
  const int wave = threadIdx.x >> 5;
  const int bh   = blockIdx.x;                // 0..63
  const int qt   = blockIdx.y * 8 + wave;     // 0..63
  const int kg   = lane >> 4;
  const int ln   = lane & 15;
  const int q0   = qt * 32;
  const float scale = 0.125f;                 // 1/sqrt(64)

  v16bf aq[2][2];                             // [mi][d-half]
#pragma unroll
  for (int mi = 0; mi < 2; ++mi) {
    const __bf16* qrow = Qb + ((size_t)bh * SS + q0 + mi * 16 + ln) * DK;
    aq[mi][0] = load_afrag(qrow, kg);         // d = 0..31
    aq[mi][1] = load_afrag(qrow + 32, kg);    // d = 32..63
  }

  float m[2][8], l[2][8];
  v8f ctx[2][4] = {};
#pragma unroll
  for (int mi = 0; mi < 2; ++mi)
#pragma unroll
    for (int r = 0; r < 8; ++r) { m[mi][r] = -INFINITY; l[mi][r] = 0.0f; }

  const int nkt = qt + 1;                     // causal: keys <= q0+31
  for (int kt = 0; kt < nkt; ++kt) {
    // ---- scores: 32 x 32 = four 16x16 C frags, K(d)=64 in two steps
    v8f sc[2][2] = {};
#pragma unroll
    for (int n = 0; n < 2; ++n) {
      const __bf16* krow = Kb + ((size_t)bh * SS + (kt * 32 + n * 16 + ln)) * DK;
      v16bf b0 = *(const v16bf*)(krow + 16 * kg);       // d = 0..31
      v16bf b1 = *(const v16bf*)(krow + 32 + 16 * kg);  // d = 32..63
#pragma unroll
      for (int mi = 0; mi < 2; ++mi) {
        sc[mi][n] = __builtin_amdgcn_wmma_f32_16x16x32_bf16(
            false, aq[mi][0], false, b0, (short)0, sc[mi][n], false, false);
        sc[mi][n] = __builtin_amdgcn_wmma_f32_16x16x32_bf16(
            false, aq[mi][1], false, b1, (short)0, sc[mi][n], false, false);
      }
    }

    // ---- causal mask + online softmax per 16-row sub-tile
#pragma unroll
    for (int mi = 0; mi < 2; ++mi) {
#pragma unroll
      for (int r = 0; r < 8; ++r) {
        const int rowg = q0 + mi * 16 + r + 8 * kg;
        float v0 = sc[mi][0][r], v1 = sc[mi][1][r];
        v0 = ((kt * 32 + ln)      <= rowg) ? v0 * scale : -1e30f;
        v1 = ((kt * 32 + 16 + ln) <= rowg) ? v1 * scale : -1e30f;
        float mx = fmaxf(v0, v1);
#pragma unroll
        for (int off = 8; off >= 1; off >>= 1)
          mx = fmaxf(mx, __shfl_xor(mx, off, 32)); // within 16-lane half
        const float mn = fmaxf(m[mi][r], mx);
        const float alpha = __expf(m[mi][r] - mn);
        const float p0 = __expf(v0 - mn);
        const float p1 = __expf(v1 - mn);
        float ps = p0 + p1;
#pragma unroll
        for (int off = 8; off >= 1; off >>= 1)
          ps += __shfl_xor(ps, off, 32);
        l[mi][r] = l[mi][r] * alpha + ps;
        m[mi][r] = mn;
#pragma unroll
        for (int t = 0; t < 4; ++t) ctx[mi][t][r] *= alpha;
        plds[wave][mi * 16 + r + 8 * kg][ln]      = (__bf16)p0;
        plds[wave][mi * 16 + r + 8 * kg][16 + ln] = (__bf16)p1;
      }
    }
    asm volatile("s_wait_dscnt 0" ::: "memory");   // wave-private LDS RAW

    // ---- reload P in A-fragment layout; PV: ctx(32x64) += P(32x32)xV(32x64)
    v16bf pa[2];
#pragma unroll
    for (int mi = 0; mi < 2; ++mi)
      pa[mi] = load_afrag(&plds[wave][mi * 16 + ln][0], kg);
#pragma unroll
    for (int t = 0; t < 4; ++t) {
      v16bf bv = *(const v16bf*)(Vt + ((size_t)bh * DK + (t * 16 + ln)) * SS
                                    + kt * 32 + 16 * kg);
#pragma unroll
      for (int mi = 0; mi < 2; ++mi)
        ctx[mi][t] = __builtin_amdgcn_wmma_f32_16x16x32_bf16(
            false, pa[mi], false, bv, (short)0, ctx[mi][t], false, false);
    }
  }

  // ---- epilogue: normalize and write bf16 concat [b, s, h*dk]
  const int b = bh >> 4, h = bh & 15;
#pragma unroll
  for (int mi = 0; mi < 2; ++mi)
#pragma unroll
    for (int r = 0; r < 8; ++r) {
      const float inv = 1.0f / l[mi][r];
      const int row = q0 + mi * 16 + r + 8 * kg;
#pragma unroll
      for (int t = 0; t < 4; ++t)
        Cc[((size_t)b * SS + row) * DD + h * DK + t * 16 + ln] =
            (__bf16)(ctx[mi][t][r] * inv);
    }
}

// ---------------------------------------------------------------------------
// Kernel 3: output projection. out[bs, d] = sum_f concat[bs, f] * Wo[d, f]
// One wave per 64x64 tile (16 WMMAs per k-step), fp32 stores.
// ---------------------------------------------------------------------------
__global__ __launch_bounds__(256)
void out_proj_kernel(const __bf16* __restrict__ Cc, const __bf16* __restrict__ Wob,
                     float* __restrict__ out) {
  const int lane = threadIdx.x & 31;
  const int wave = threadIdx.x >> 5;
  const int gid  = blockIdx.x * 8 + wave;     // 0..2047
  const int mt   = gid & 127;                 // 128 M tiles (B*S/64)
  const int nt   = gid >> 7;                  // 16 N tiles (D/64)
  const int kg   = lane >> 4, ln = lane & 15;
  const int m0   = mt * 64, n0 = nt * 64;

  v8f acc[4][4] = {};
  const __bf16* abase = Cc + (size_t)(m0 + ln) * DD;
  for (int k0 = 0; k0 < DD; k0 += 32) {
    v16bf a[4];
#pragma unroll
    for (int mi = 0; mi < 4; ++mi)
      a[mi] = load_afrag(abase + (size_t)mi * 16 * DD + k0, kg);
#pragma unroll
    for (int t = 0; t < 4; ++t) {
      v16bf bf = *(const v16bf*)(Wob + (size_t)(n0 + t * 16 + ln) * DD + k0 + 16 * kg);
#pragma unroll
      for (int mi = 0; mi < 4; ++mi)
        acc[mi][t] = __builtin_amdgcn_wmma_f32_16x16x32_bf16(
            false, a[mi], false, bf, (short)0, acc[mi][t], false, false);
    }
  }
  float* dst = out + (size_t)(m0 + 8 * kg) * DD + n0 + ln;
#pragma unroll
  for (int mi = 0; mi < 4; ++mi)
#pragma unroll
    for (int t = 0; t < 4; ++t)
#pragma unroll
      for (int r = 0; r < 8; ++r)
        dst[(size_t)(mi * 16 + r) * DD + t * 16] = acc[mi][t][r];
}

// ---------------------------------------------------------------------------
extern "C" void kernel_launch(void* const* d_in, const int* in_sizes, int n_in,
                              void* d_out, int out_size, void* d_ws, size_t ws_size,
                              hipStream_t stream) {
  const float* x    = (const float*)d_in[0];   // [B,S,D]
  const float* Wqkv = (const float*)d_in[1];   // [3,H,dk,D]
  const float* Wo   = (const float*)d_in[2];   // [D,H*dk]
  float* out = (float*)d_out;                  // [B,S,D] fp32

  const size_t nX  = (size_t)BB * SS * DD;     // 8,388,608
  const size_t nW  = (size_t)3 * HH * DK * DD; // 3,145,728
  const size_t nWo = (size_t)DD * DD;          // 1,048,576
  const size_t nQ  = (size_t)BB * HH * SS * DK;// 8,388,608

  __bf16* Qb  = (__bf16*)d_ws;                 // 16 MB
  __bf16* Kb  = Qb + nQ;                       // 16 MB
  __bf16* Vt  = Kb + nQ;                       // 16 MB  ([bh, dk, S])
  __bf16* Cc  = Vt + nQ;                       // 16 MB  (bf16 concat)
  __bf16* xb  = Cc + nQ;                       // 16 MB
  __bf16* Wb  = xb + nX;                       //  6 MB
  __bf16* Wob = Wb + nW;                       //  2 MB

  convert_kernel<<<(int)(nX  / 4 / 256), 256, 0, stream>>>(x, xb, (int)nX);
  convert_kernel<<<(int)(nW  / 4 / 256), 256, 0, stream>>>(Wqkv, Wb, (int)nW);
  convert_kernel<<<(int)(nWo / 4 / 256), 256, 0, stream>>>(Wo, Wob, (int)nWo);

  // 3*B*H*(S/64) waves = 6144 -> 768 blocks of 8 waves
  qkv_proj_kernel<<<768, 256, 0, stream>>>(xb, Wb, Qb, Kb, Vt);
  // (B*H) x (S/32/8) blocks, 8 waves each
  attn_kernel<<<dim3(BB * HH, SS / 32 / 8), 256, 0, stream>>>(Qb, Kb, Vt, Cc);
  // (B*S/64)*(D/64) waves = 2048 -> 256 blocks
  out_proj_kernel<<<256, 256, 0, stream>>>(Cc, Wob, out);
}
